// GNN_6511170420958
// MI455X (gfx1250) — compile-verified, matrix operand order checked by reading, not run on hardware
//
#include <hip/hip_runtime.h>
#include <hip/hip_bf16.h>

#define N_NODES 131072
#define N_EDGES 1048576
#define HDIM    128
#define H3      384
#define H6      256
#define N_PROB  64

typedef __attribute__((ext_vector_type(16))) _Float16 v16h;
typedef __attribute__((ext_vector_type(8)))  float    v8f;

// ---------------------------------------------------------------------------
// Generic zero-fill
__global__ void zero_f32(float* __restrict__ p, long n) {
    long i = (long)blockIdx.x * blockDim.x + threadIdx.x;
    if (i < n) p[i] = 0.0f;
}

// ---------------------------------------------------------------------------
// emb1: x = relu(G @ W1 + b1), G [N,8], W1 [8,128]
__global__ void emb1_kernel(const float* __restrict__ G, const float* __restrict__ W1,
                            const float* __restrict__ b1, float* __restrict__ out) {
    long idx = (long)blockIdx.x * blockDim.x + threadIdx.x;   // N*128 threads
    long n = idx >> 7;
    int  j = (int)(idx & 127);
    if (n >= N_NODES) return;
    const float* g = G + n * 8;
    float acc = b1[j];
#pragma unroll
    for (int k = 0; k < 8; ++k) acc += g[k] * W1[k * HDIM + j];
    out[n * HDIM + j] = fmaxf(acc, 0.0f);
}

// ---------------------------------------------------------------------------
// WMMA GEMM: Out[N,Nout] = act(A[N,K] @ W + bias)
//   TRANSW == 0: W is [K, Nout] row-major,  W(k,n) = W[k*Nout + n]
//   TRANSW == 1: W is [Nout, K] row-major (PyTorch Wih/Whh), W(k,n) = W[n*K + k]
//
// Block = 256 threads (8 waves). Each wave owns a 16-row x 128-col stripe
// (8 accumulators) so A is read once per 128-col group (gridDim.y groups).
// The block's ENTIRE weight panel (K x 128, converted to f16, laid out in
// exact WMMA B lane order) is staged into dynamic LDS once before the k-loop:
// thread (ct=wave, lane) writes, for every k-step, the 16-half vector that
// lane `lane` needs for col-tile `ct`. A single __syncthreads later, the
// k-loop is a pure stream of global A loads + ds_load_b128 + v_wmma with no
// further synchronization.
template <int TRANSW>
__global__ __launch_bounds__(256) void gemm_wmma(
        const float* __restrict__ A, const float* __restrict__ W,
        const float* __restrict__ bias, float* __restrict__ Out,
        int K, int Nout, int act) {
    extern __shared__ _Float16 Bs[];          // (K/32) * 8 tiles * 32 lanes * 16 halves

    int tid  = threadIdx.x;
    int lane = tid & 31;
    int wave = tid >> 5;

    int r  = lane & 15;
    int kb = (lane < 16) ? 0 : 8;             // 16-bit A/B lane layout: K-base split
    long rowbase = (long)blockIdx.x * 128 + wave * 16;
    long row     = rowbase + r;               // A row owned by this lane
    int  colbase = blockIdx.y * 128;
    int  fcol    = colbase + wave * 16 + r;   // W column this thread stages

    int ksteps = K >> 5;

    // ---- stage full weight panel into LDS (f16, WMMA B lane layout) ----
    for (int ks = 0; ks < ksteps; ++ks) {
        int k0 = ks * 32;
        v16h b;
        if (TRANSW) {
            const float* Wp = W + (long)fcol * K + k0 + kb;   // contiguous in k
#pragma unroll
            for (int i = 0; i < 8; ++i) {
                b[i]     = (_Float16)Wp[i];
                b[8 + i] = (_Float16)Wp[16 + i];
            }
        } else {
            const float* Wp = W + (long)(k0 + kb) * Nout + fcol;
#pragma unroll
            for (int i = 0; i < 8; ++i) {
                b[i]     = (_Float16)Wp[(long)i * Nout];
                b[8 + i] = (_Float16)Wp[(long)(16 + i) * Nout];
            }
        }
        *(v16h*)&Bs[(ks * 256 + wave * 32 + lane) * 16] = b;
    }

    v8f acc[8];
#pragma unroll
    for (int ct = 0; ct < 8; ++ct) acc[ct] = (v8f){};

    __syncthreads();                          // whole weight panel visible

    // ---- main loop: A stream + LDS B + WMMA, no barriers ----
    for (int ks = 0; ks < ksteps; ++ks) {
        int k0 = ks * 32;
        v16h a;
        const float* Ap = A + row * (long)K + k0 + kb;
        __builtin_prefetch(Ap + 32, 0, 1);    // next k-tile -> global_prefetch_b8
#pragma unroll
        for (int i = 0; i < 8; ++i) {         // a[i]<->K=kb+i, a[8+i]<->K=16+kb+i
            a[i]     = (_Float16)Ap[i];
            a[8 + i] = (_Float16)Ap[16 + i];
        }
#pragma unroll
        for (int ct = 0; ct < 8; ++ct) {
            v16h b = *(const v16h*)&Bs[(ks * 256 + ct * 32 + lane) * 16];
            acc[ct] = __builtin_amdgcn_wmma_f32_16x16x32_f16(
                          false, a, false, b, (short)0, acc[ct], false, false);
        }
    }

    int mbase = (lane < 16) ? 0 : 8;          // C layout: VGPR v <-> M = mbase+v
#pragma unroll
    for (int ct = 0; ct < 8; ++ct) {
        int col = colbase + ct * 16 + r;
        float bv = bias ? bias[col] : 0.0f;
#pragma unroll
        for (int v = 0; v < 8; ++v) {
            long m = rowbase + mbase + v;
            float val = acc[ct][v] + bv;
            if (act) val = fmaxf(val, 0.0f);
            Out[m * (long)Nout + col] = val;
        }
    }
}

// ---------------------------------------------------------------------------
// Edge message + scatter-add: agg[dst] += t[src] * ew   (32 threads per edge)
__global__ void edge_scatter(const float* __restrict__ t, const int* __restrict__ src,
                             const int* __restrict__ dst, const float* __restrict__ ew,
                             float* __restrict__ agg) {
    long gid = (long)blockIdx.x * blockDim.x + threadIdx.x;
    long e = gid >> 5;
    int  c = (int)(gid & 31) * 4;
    if (e >= N_EDGES) return;
    int s = src[e], d = dst[e];
    float w = ew[e];
    const float4 tv = *(const float4*)(t + (long)s * HDIM + c);
    float* ap = agg + (long)d * HDIM + c;
    atomicAdd(ap + 0, tv.x * w);
    atomicAdd(ap + 1, tv.y * w);
    atomicAdd(ap + 2, tv.z * w);
    atomicAdd(ap + 3, tv.w * w);
}

// ---------------------------------------------------------------------------
// GRU cell (PyTorch semantics), in-place on h; optional conv-level ReLU.
__device__ __forceinline__ float sigmoidf(float x) { return 1.0f / (1.0f + __expf(-x)); }

__global__ void gru_cell(const float* __restrict__ gi, const float* __restrict__ gh,
                         float* __restrict__ h, int relu_out) {
    long idx = (long)blockIdx.x * blockDim.x + threadIdx.x;   // N*128
    long n = idx >> 7;
    int  j = (int)(idx & 127);
    if (n >= N_NODES) return;
    const float* gir = gi + n * (long)H3;
    const float* ghr = gh + n * (long)H3;
    float i_r = gir[j], i_z = gir[HDIM + j], i_n = gir[2 * HDIM + j];
    float h_r = ghr[j], h_z = ghr[HDIM + j], h_n = ghr[2 * HDIM + j];
    float r = sigmoidf(i_r + h_r);
    float z = sigmoidf(i_z + h_z);
    float nn = tanhf(i_n + r * h_n);
    float hv = h[idx];
    float out = (1.0f - z) * nn + z * hv;
    if (relu_out) out = fmaxf(out, 0.0f);
    h[idx] = out;
}

// ---------------------------------------------------------------------------
// Segment sums over sorted segment_ids: per-block run-length local accumulation.
__global__ void seg_sum(const float* __restrict__ x, const int* __restrict__ seg,
                        float* __restrict__ sums, float* __restrict__ counts) {
    int j = threadIdx.x;                       // 128 threads (one per feature)
    long n0 = (long)blockIdx.x * 128;          // 128 nodes per block
    int cur = seg[n0];
    float acc = 0.0f; int cnt = 0;
    for (int i = 0; i < 128; ++i) {
        long n = n0 + i;
        int s = seg[n];
        if (s != cur) {
            atomicAdd(&sums[(long)cur * HDIM + j], acc);
            if (j == 0) atomicAdd(&counts[cur], (float)cnt);
            acc = 0.0f; cnt = 0; cur = s;
        }
        acc += x[n * HDIM + j]; cnt++;
    }
    atomicAdd(&sums[(long)cur * HDIM + j], acc);
    if (j == 0) atomicAdd(&counts[cur], (float)cnt);
}

__global__ void seg_mean(const float* __restrict__ sums, const float* __restrict__ counts,
                         float* __restrict__ gmean) {
    int s = blockIdx.x, j = threadIdx.x;       // 64 x 128
    gmean[(long)s * HDIM + j] = sums[(long)s * HDIM + j] / fmaxf(counts[s], 1.0f);
}

// ---------------------------------------------------------------------------
// u = concat([x, gmean[seg]], -1)  -> [N, 256]
__global__ void concat_kernel(const float* __restrict__ x, const float* __restrict__ gmean,
                              const int* __restrict__ seg, float* __restrict__ u) {
    long idx = (long)blockIdx.x * blockDim.x + threadIdx.x;   // N*256
    long n = idx >> 8;
    int  j = (int)(idx & 255);
    if (n >= N_NODES) return;
    float v = (j < HDIM) ? x[n * HDIM + j]
                         : gmean[(long)seg[n] * HDIM + (j - HDIM)];
    u[n * (long)H6 + j] = v;
}

// ---------------------------------------------------------------------------
// Final [N,128] @ [128,1] + b
__global__ void final_lin(const float* __restrict__ u3, const float* __restrict__ W,
                          const float* __restrict__ b, float* __restrict__ out) {
    long n = (long)blockIdx.x * blockDim.x + threadIdx.x;
    if (n >= N_NODES) return;
    const float* row = u3 + n * HDIM;
    float acc = 0.0f;
#pragma unroll 8
    for (int k = 0; k < HDIM; ++k) acc += row[k] * W[k];
    out[n] = acc + b[0];
}

// ---------------------------------------------------------------------------
static inline void launch_gemm(const float* A, const float* W, const float* bias,
                               float* Out, int K, int Nout, int transW,
                               int act, hipStream_t stream) {
    dim3 grid(N_NODES / 128, Nout / 128);      // Nout in {128,256,384}: exact
    size_t lds = (size_t)(K >> 5) * 8 * 32 * 16 * sizeof(_Float16);  // K*256 B
    if (transW)
        gemm_wmma<1><<<grid, 256, lds, stream>>>(A, W, bias, Out, K, Nout, act);
    else
        gemm_wmma<0><<<grid, 256, lds, stream>>>(A, W, bias, Out, K, Nout, act);
}

extern "C" void kernel_launch(void* const* d_in, const int* in_sizes, int n_in,
                              void* d_out, int out_size, void* d_ws, size_t ws_size,
                              hipStream_t stream) {
    const float* G    = (const float*)d_in[0];
    const int*   eidx = (const int*)  d_in[1];
    const int*   src  = eidx;
    const int*   dst  = eidx + N_EDGES;
    const float* ew   = (const float*)d_in[2];
    const int*   seg  = (const int*)  d_in[4];
    const float* W_emb1 = (const float*)d_in[5];  const float* b_emb1 = (const float*)d_in[6];
    const float* W_emb2 = (const float*)d_in[7];  const float* b_emb2 = (const float*)d_in[8];
    const float* conv_W[2]   = { (const float*)d_in[9],  (const float*)d_in[14] };
    const float* conv_Wih[2] = { (const float*)d_in[10], (const float*)d_in[15] };
    const float* conv_Whh[2] = { (const float*)d_in[11], (const float*)d_in[16] };
    const float* conv_bih[2] = { (const float*)d_in[12], (const float*)d_in[17] };
    const float* conv_bhh[2] = { (const float*)d_in[13], (const float*)d_in[18] };
    const float* W_g1 = (const float*)d_in[19]; const float* b_g1 = (const float*)d_in[20];
    const float* W_g2 = (const float*)d_in[21]; const float* b_g2 = (const float*)d_in[22];
    const float* W_l  = (const float*)d_in[23]; const float* b_l  = (const float*)d_in[24];
    const float* W_l2 = (const float*)d_in[25]; const float* b_l2 = (const float*)d_in[26];
    const float* W_l3 = (const float*)d_in[27]; const float* b_l3 = (const float*)d_in[28];
    float* out = (float*)d_out;

    const long NB = (long)N_NODES * HDIM;     // elements per [N,128] buffer
    const long GB = (long)N_NODES * H3;       // elements per [N,384] buffer
    char* ws = (char*)d_ws;
    float* h    = (float*)(ws);
    float* t    = (float*)(ws + NB * 4);
    float* agg  = (float*)(ws + 2 * NB * 4);
    float* gi   = (float*)(ws + 3 * NB * 4);
    float* gh   = (float*)(ws + 3 * NB * 4 + GB * 4);
    float* sums = (float*)(ws + 3 * NB * 4 + 2 * GB * 4);
    float* counts = sums + N_PROB * HDIM;
    float* gmean  = counts + N_PROB;
    float* u  = gi;                           // [N,256] fits in [N,384] region
    float* u2 = gh;
    float* xe = agg;
    float* x1 = t;
    float* u3 = t;

    const int TPB = 256;
    const int BLK_NH  = (int)(NB / TPB);                  // N*128 threads
    const int BLK_E   = (int)((long)N_EDGES * 32 / TPB);  // E*32 threads
    const int BLK_N2H = (int)((long)N_NODES * H6 / TPB);  // N*256 threads

    // ---- embedding: t = relu(G@W1+b1); h = relu(t@W2+b2) ----
    emb1_kernel<<<BLK_NH, TPB, 0, stream>>>(G, W_emb1, b_emb1, t);
    launch_gemm(t, W_emb2, b_emb2, h, HDIM, HDIM, 0, 1, stream);

    // ---- two GatedGraphConv layers, 2 inner steps each ----
    for (int conv = 0; conv < 2; ++conv) {
        for (int l = 0; l < 2; ++l) {
            launch_gemm(h, conv_W[conv] + (long)l * HDIM * HDIM, nullptr, t,
                        HDIM, HDIM, 0, 0, stream);
            zero_f32<<<BLK_NH, TPB, 0, stream>>>(agg, NB);
            edge_scatter<<<BLK_E, TPB, 0, stream>>>(t, src, dst, ew, agg);
            launch_gemm(agg, conv_Wih[conv], conv_bih[conv], gi,
                        HDIM, H3, 1, 0, stream);
            launch_gemm(h, conv_Whh[conv], conv_bhh[conv], gh,
                        HDIM, H3, 1, 0, stream);
            gru_cell<<<BLK_NH, TPB, 0, stream>>>(gi, gh, h, (l == 1) ? 1 : 0);
        }
    }

    // ---- head: x1 = relu(h@Wg1+b); xe = x1@Wg2+b ----
    launch_gemm(h, W_g1, b_g1, x1, HDIM, HDIM, 0, 1, stream);
    launch_gemm(x1, W_g2, b_g2, xe, HDIM, HDIM, 0, 0, stream);

    // ---- per-problem segment mean ----
    {
        long zn = (long)N_PROB * HDIM + N_PROB;
        zero_f32<<<(int)((zn + TPB - 1) / TPB), TPB, 0, stream>>>(sums, zn);
    }
    seg_sum<<<N_NODES / 128, 128, 0, stream>>>(xe, seg, sums, counts);
    seg_mean<<<N_PROB, HDIM, 0, stream>>>(sums, counts, gmean);
    concat_kernel<<<BLK_N2H, TPB, 0, stream>>>(xe, gmean, seg, u);

    // ---- MLP head ----
    launch_gemm(u,  W_l,  b_l,  u2, H6, H6,   0, 1, stream);
    launch_gemm(u2, W_l2, b_l2, u3, H6, HDIM, 0, 1, stream);
    final_lin<<<N_NODES / TPB, TPB, 0, stream>>>(u3, W_l3, b_l3, out);
}
